// BertSelfAttention_69724499083363
// MI455X (gfx1250) — compile-verified, matrix operand order checked by reading, not run on hardware
//
#include <hip/hip_runtime.h>

#define B_ 8
#define S_ 1024
#define D_ 1024
#define H_ 16
#define HD_ 64

typedef __attribute__((ext_vector_type(16))) __bf16 v16bf;
typedef __attribute__((ext_vector_type(2))) __bf16 v2bf;
typedef __attribute__((ext_vector_type(8))) float v8f;

union Frag16 {
  v16bf v;
  uint4 q[2];
};

// Hardware f32 -> bf16 (RNE); lowers to v_cvt ops on gfx1250.
__device__ __forceinline__ unsigned short f2bf(float f) {
  __bf16 b = (__bf16)f;
  return __builtin_bit_cast(unsigned short, b);
}

// Packed pair -> one v_cvt_pk_bf16_f32.
__device__ __forceinline__ unsigned int pack2_bf16(float a, float b) {
  v2bf t;
  t.x = (__bf16)a;
  t.y = (__bf16)b;
  return __builtin_bit_cast(unsigned int, t);
}

__device__ __forceinline__ v8f wmma_bf16(const Frag16& a, const Frag16& b, v8f c) {
  // (neg_a, A, neg_b, B, c_mod, C, reuse_a, reuse_b)
  return __builtin_amdgcn_wmma_f32_16x16x32_bf16(false, a.v, false, b.v,
                                                 (short)0, c, false, false);
}

// ---------------------------------------------------------------------------
// Kernel 1: fused QKV projection GEMM.  X[8192x1024] * W[1024x1024] + b
// blockIdx.z selects (Wq,bq,Q) / (Wk,bk,K) / (Wv,bv,V).
// Output written bf16, head-major [B,H,S,HD] into workspace.
// Block: 256 threads = 8 waves, tile 128(M) x 64(N), BK=32.
// Each wave: 32x32 sub-tile = 2x2 WMMA 16x16 tiles.
// ---------------------------------------------------------------------------
__global__ __launch_bounds__(256) void qkv_gemm(
    const float* __restrict__ X,
    const float* __restrict__ Wq, const float* __restrict__ bq,
    const float* __restrict__ Wk, const float* __restrict__ bk,
    const float* __restrict__ Wv, const float* __restrict__ bv,
    unsigned short* __restrict__ q_ws, unsigned short* __restrict__ k_ws,
    unsigned short* __restrict__ v_ws) {
  const float* W;
  const float* bias;
  unsigned short* O;
  if (blockIdx.z == 0) { W = Wq; bias = bq; O = q_ws; }
  else if (blockIdx.z == 1) { W = Wk; bias = bk; O = k_ws; }
  else { W = Wv; bias = bv; O = v_ws; }

  // row stride 48 bf16 = 96 bytes (16B-aligned rows, padded vs. 64B)
  __shared__ __align__(16) unsigned short As[128 * 48];  // X tile (bf16)
  __shared__ __align__(16) unsigned short Bs[64 * 48];   // W^T tile (bf16)

  const int tid = threadIdx.x;
  const int wave = tid >> 5;
  const int lane = tid & 31;
  const int m = lane & 15;
  const int half = lane >> 4;
  const int waveM = (wave >> 1) * 32;
  const int waveN = (wave & 1) * 32;
  const int blockM = blockIdx.x * 128;
  const int blockN = blockIdx.y * 64;

  v8f c00 = {}, c01 = {}, c10 = {}, c11 = {};

  for (int k0 = 0; k0 < D_; k0 += 32) {
    // ---- batched loads first (forms a VMEM clause, single wait) ----
    float4 fa[4];
    for (int i = 0; i < 4; ++i) {
      int j = tid + i * 256;
      int row = j >> 3;
      int kq = (j & 7) << 2;
      fa[i] = *(const float4*)(X + (size_t)(blockM + row) * D_ + k0 + kq);
    }
    float4 fw[2];
    for (int i = 0; i < 2; ++i) {
      int j = tid + i * 256;
      int k = j >> 4;
      int nq = (j & 15) << 2;
      fw[i] = *(const float4*)(W + (size_t)(k0 + k) * D_ + blockN + nq);
    }
    // speculative prefetch of next k-step (translation faults dropped silently)
    __builtin_prefetch(X + (size_t)(blockM + (tid >> 3)) * D_ + k0 + 32 +
                           ((tid & 7) << 2), 0, 0);
    __builtin_prefetch(W + (size_t)(k0 + 32 + (tid >> 4)) * D_ + blockN +
                           ((tid & 15) << 2), 0, 0);

    // ---- convert + store to LDS ----
    for (int i = 0; i < 4; ++i) {
      int j = tid + i * 256;
      int row = j >> 3;
      int kq = (j & 7) << 2;
      uint2 d;
      d.x = pack2_bf16(fa[i].x, fa[i].y);
      d.y = pack2_bf16(fa[i].z, fa[i].w);
      *(uint2*)&As[row * 48 + kq] = d;
    }
    for (int i = 0; i < 2; ++i) {
      int j = tid + i * 256;
      int k = j >> 4;
      int nq = (j & 15) << 2;
      Bs[(nq + 0) * 48 + k] = f2bf(fw[i].x);
      Bs[(nq + 1) * 48 + k] = f2bf(fw[i].y);
      Bs[(nq + 2) * 48 + k] = f2bf(fw[i].z);
      Bs[(nq + 3) * 48 + k] = f2bf(fw[i].w);
    }
    __syncthreads();

    // gather WMMA fragments (A layout: lane row = m, K = {8h..8h+7} U {16+8h..})
    Frag16 a0, a1, b0, b1;
    {
      const uint4* p = (const uint4*)&As[(waveM + m) * 48];
      a0.q[0] = p[half]; a0.q[1] = p[2 + half];
      p = (const uint4*)&As[(waveM + 16 + m) * 48];
      a1.q[0] = p[half]; a1.q[1] = p[2 + half];
      p = (const uint4*)&Bs[(waveN + m) * 48];
      b0.q[0] = p[half]; b0.q[1] = p[2 + half];
      p = (const uint4*)&Bs[(waveN + 16 + m) * 48];
      b1.q[0] = p[half]; b1.q[1] = p[2 + half];
    }
    c00 = wmma_bf16(a0, b0, c00);
    c01 = wmma_bf16(a0, b1, c01);
    c10 = wmma_bf16(a1, b0, c10);
    c11 = wmma_bf16(a1, b1, c11);
    __syncthreads();
  }

  // epilogue: bias add, convert bf16, write head-major [B,H,S,HD]
  for (int tm = 0; tm < 2; ++tm) {
    for (int tn = 0; tn < 2; ++tn) {
      v8f c = (tm == 0) ? (tn == 0 ? c00 : c01) : (tn == 0 ? c10 : c11);
      int col = blockN + waveN + tn * 16 + m;
      int hh = col >> 6;
      int hd = col & 63;
      float bsv = bias[col];
      for (int r = 0; r < 8; ++r) {
        int row = blockM + waveM + tm * 16 + r + 8 * half;
        int bb = row >> 10;
        int ss = row & 1023;
        O[((size_t)(bb * H_ + hh) * S_ + ss) * HD_ + hd] = f2bf(c[r] + bsv);
      }
    }
  }
}

// ---------------------------------------------------------------------------
// Kernel 2: fused flash-style attention per (b, h).
// Block = 256 threads = 8 waves; block covers 128 q rows; wave owns 16 q rows.
// Streams keys in chunks of 32 with online softmax; never materializes SxS.
// ---------------------------------------------------------------------------
__global__ __launch_bounds__(256) void attn_fused(
    const unsigned short* __restrict__ q_ws,
    const unsigned short* __restrict__ k_ws,
    const unsigned short* __restrict__ v_ws,
    const float* __restrict__ amask,
    float* __restrict__ out) {
  __shared__ __align__(16) unsigned short Kt[32 * 80];     // [key][hd], padded rows
  __shared__ __align__(16) unsigned short Vt[64 * 48];     // [hd][key], padded rows
  __shared__ __align__(16) unsigned short Pb[8][16 * 32];  // per-wave P scratch

  const int tid = threadIdx.x;
  const int wave = tid >> 5;
  const int lane = tid & 31;
  const int m = lane & 15;
  const int half = lane >> 4;
  const int h = blockIdx.y;
  const int b = blockIdx.z;
  const int qbase = blockIdx.x * 128 + wave * 16;

  const size_t headOff = ((size_t)(b * H_ + h)) * S_ * HD_;

  // Q fragments for this wave's 16 rows (HD=64 => two K=32 A-fragments)
  Frag16 qa0, qa1;
  {
    const uint4* qp = (const uint4*)(q_ws + headOff + (size_t)(qbase + m) * HD_);
    qa0.q[0] = qp[half];     qa0.q[1] = qp[2 + half];
    qa1.q[0] = qp[4 + half]; qa1.q[1] = qp[6 + half];
  }

  float mrow[8], lrow[8];
  v8f acc0 = {}, acc1 = {}, acc2 = {}, acc3 = {};
  for (int r = 0; r < 8; ++r) { mrow[r] = -1e30f; lrow[r] = 0.0f; }

  const float scale = 0.125f;  // 1/sqrt(HD)

  for (int kb = 0; kb < S_; kb += 32) {
    // cooperative staging: K chunk [32][64] and V chunk transposed [64][32]
    {
      int key = tid >> 3;
      int hq = (tid & 7) << 3;
      const unsigned short* ksrc = k_ws + headOff + (size_t)(kb + key) * HD_ + hq;
      const unsigned short* vsrc = v_ws + headOff + (size_t)(kb + key) * HD_ + hq;
      uint4 kv = *(const uint4*)ksrc;   // batched loads, single wait
      uint4 vv = *(const uint4*)vsrc;
      __builtin_prefetch(ksrc + 32 * HD_, 0, 0);  // speculative, next chunk
      __builtin_prefetch(vsrc + 32 * HD_, 0, 0);
      *(uint4*)&Kt[key * 80 + hq] = kv;
      const unsigned short* pv = (const unsigned short*)&vv;
      for (int u = 0; u < 8; ++u) Vt[(hq + u) * 48 + key] = pv[u];
    }
    __syncthreads();

    // scores: two 16x16 tiles (keys kb..kb+15, kb+16..kb+31)
    v8f s0 = {}, s1 = {};
    {
      Frag16 klo, khi;
      const uint4* kp = (const uint4*)&Kt[(m) * 80];
      klo.q[0] = kp[half];     klo.q[1] = kp[2 + half];
      khi.q[0] = kp[4 + half]; khi.q[1] = kp[6 + half];
      s0 = wmma_bf16(qa0, klo, s0);
      s0 = wmma_bf16(qa1, khi, s0);
      kp = (const uint4*)&Kt[(16 + m) * 80];
      klo.q[0] = kp[half];     klo.q[1] = kp[2 + half];
      khi.q[0] = kp[4 + half]; khi.q[1] = kp[6 + half];
      s1 = wmma_bf16(qa0, klo, s1);
      s1 = wmma_bf16(qa1, khi, s1);
    }
    float mk0 = amask[(size_t)b * S_ + kb + m];
    float mk1 = amask[(size_t)b * S_ + kb + 16 + m];

    // online softmax: rows live across 16 lanes of each half-wave
    for (int r = 0; r < 8; ++r) {
      float v0 = s0[r] * scale + mk0;
      float v1 = s1[r] * scale + mk1;
      float cm = fmaxf(v0, v1);
      for (int off = 1; off < 16; off <<= 1)
        cm = fmaxf(cm, __shfl_xor(cm, off, 32));
      float newm = fmaxf(mrow[r], cm);
      float so = __expf(mrow[r] - newm);
      float p0 = __expf(v0 - newm);
      float p1 = __expf(v1 - newm);
      float sm = p0 + p1;
      for (int off = 1; off < 16; off <<= 1)
        sm += __shfl_xor(sm, off, 32);
      lrow[r] = lrow[r] * so + sm;
      mrow[r] = newm;
      acc0[r] *= so; acc1[r] *= so; acc2[r] *= so; acc3[r] *= so;
      int prow = (r + 8 * half) * 32;
      Pb[wave][prow + m] = f2bf(p0);
      Pb[wave][prow + 16 + m] = f2bf(p1);
    }
    // in-wave LDS RAW: wait for DS stores before WMMA consumes P fragment
    asm volatile("s_wait_dscnt 0" ::: "memory");

    Frag16 pf;
    {
      const uint4* pp = (const uint4*)&Pb[wave][m * 32];
      pf.q[0] = pp[half]; pf.q[1] = pp[2 + half];
    }
    Frag16 vb;
    const uint4* vp;
    vp = (const uint4*)&Vt[(0 * 16 + m) * 48];
    vb.q[0] = vp[half]; vb.q[1] = vp[2 + half];
    acc0 = wmma_bf16(pf, vb, acc0);
    vp = (const uint4*)&Vt[(1 * 16 + m) * 48];
    vb.q[0] = vp[half]; vb.q[1] = vp[2 + half];
    acc1 = wmma_bf16(pf, vb, acc1);
    vp = (const uint4*)&Vt[(2 * 16 + m) * 48];
    vb.q[0] = vp[half]; vb.q[1] = vp[2 + half];
    acc2 = wmma_bf16(pf, vb, acc2);
    vp = (const uint4*)&Vt[(3 * 16 + m) * 48];
    vb.q[0] = vp[half]; vb.q[1] = vp[2 + half];
    acc3 = wmma_bf16(pf, vb, acc3);
    __syncthreads();
  }

  // epilogue: normalize and write f32 output [B,S,D], merging heads
  for (int r = 0; r < 8; ++r) {
    int qrow = qbase + r + 8 * half;
    float inv = 1.0f / lrow[r];
    float* o = out + ((size_t)b * S_ + qrow) * D_ + h * HD_ + m;
    o[0]  = acc0[r] * inv;
    o[16] = acc1[r] * inv;
    o[32] = acc2[r] * inv;
    o[48] = acc3[r] * inv;
  }
}

extern "C" void kernel_launch(void* const* d_in, const int* in_sizes, int n_in,
                              void* d_out, int out_size, void* d_ws, size_t ws_size,
                              hipStream_t stream) {
  (void)in_sizes; (void)n_in; (void)out_size; (void)ws_size;
  const float* hs   = (const float*)d_in[0];
  const float* mask = (const float*)d_in[1];
  const float* Wq   = (const float*)d_in[2];
  const float* bq   = (const float*)d_in[3];
  const float* Wk   = (const float*)d_in[4];
  const float* bk   = (const float*)d_in[5];
  const float* Wv   = (const float*)d_in[6];
  const float* bv   = (const float*)d_in[7];

  unsigned short* q_ws = (unsigned short*)d_ws;
  unsigned short* k_ws = q_ws + (size_t)B_ * S_ * D_;
  unsigned short* v_ws = k_ws + (size_t)B_ * S_ * D_;
  float* out = (float*)d_out;

  dim3 gg(64, 16, 3);  // M/128, N/64, {Q,K,V}
  qkv_gemm<<<gg, dim3(256), 0, stream>>>(hs, Wq, bq, Wk, bk, Wv, bv,
                                         q_ws, k_ws, v_ws);
  dim3 ga(S_ / 128, H_, B_);
  attn_fused<<<ga, dim3(256), 0, stream>>>(q_ws, k_ws, v_ws, mask, out);
}
